// SimpleModularValueHead_6227702579249
// MI455X (gfx1250) — matrix operand order repaired and checked by reading, not compiled, not run on hardware
//
#include <hip/hip_runtime.h>
#include <cstdint>
#include <cstddef>

// Problem constants (match reference)
#define HDIM   512
#define BATCH  256
#define DLVL   6
#define KOPS   16
#define NLEAF  32
#define RCAP   8192   // max rows per op bin at the deepest level (B*32)

typedef __attribute__((ext_vector_type(16))) __bf16 bf16x16;
typedef __attribute__((ext_vector_type(8)))  float  f32x8;

__device__ __forceinline__ unsigned short f32_to_bf16(float f) {
    union { float f; unsigned int u; } v; v.f = f;
    unsigned int r = v.u + 0x7FFFu + ((v.u >> 16) & 1u);   // round-to-nearest-even
    return (unsigned short)(r >> 16);
}
__device__ __forceinline__ float bf16_to_f32(unsigned short h) {
    union { float f; unsigned int u; } v; v.u = ((unsigned int)h) << 16; return v.f;
}

// ---------------------------------------------------------------------------
// Weight tiler: fp32 row-major (C x N) -> bf16 WMMA B-matrix tiles.
// Tile (cs,nt) = 32x16 (KxN). Per-lane layout (ISA 16-bit B, wave32):
//   lane L: n = L&15, kbase = (L<16)?0:8, elem i<8 -> k = kbase + i ; elem i>=8 -> k = 16 + kbase + (i-8).
// Stored so each lane's 16 bf16 are contiguous (32B -> one b128x2 load).
// ---------------------------------------------------------------------------
__global__ void tile_weights(const float* __restrict__ src,
                             unsigned short* __restrict__ dst,
                             int C, int N, long total) {
    long idx = (long)blockIdx.x * blockDim.x + threadIdx.x;
    if (idx >= total) return;
    long perMat = (long)C * N;
    int  mat = (int)(idx / perMat);
    int  rem = (int)(idx % perMat);
    int  i    = rem & 15;
    int  lane = (rem >> 4) & 31;
    int  tile = rem >> 9;
    int  ntTiles = N >> 4;
    int  nt = tile % ntTiles;
    int  cs = tile / ntTiles;
    int  n  = nt * 16 + (lane & 15);
    int  kb = (lane < 16) ? 0 : 8;
    int  c  = cs * 32 + kb + i + ((i >= 8) ? 8 : 0);
    dst[idx] = f32_to_bf16(src[(long)mat * perMat + (long)c * N + n]);
}

// Gather leaf embeddings -> bf16 enc buffer (B x 64 x H)
__global__ void gather_leaves(const float* __restrict__ leaf_table,
                              const int* __restrict__ leaf_ids,
                              unsigned short* __restrict__ enc) {
    long idx = (long)blockIdx.x * blockDim.x + threadIdx.x;
    if (idx >= (long)BATCH * 64 * HDIM) return;
    int  c  = (int)(idx % HDIM);
    long bl = idx / HDIM;                 // b*64 + leaf position
    int  lf = leaf_ids[bl];
    enc[idx] = f32_to_bf16(leaf_table[(long)lf * HDIM + c]);
}

// ---- binning: group node-rows by op id ----
__global__ void bin_zero(int* __restrict__ counts) {
    if (threadIdx.x < KOPS) counts[threadIdx.x] = 0;
}

__global__ void bin_rows(const int* __restrict__ op_ids, int nodes,
                         int* __restrict__ counts, int* __restrict__ rowidx) {
    int r = blockIdx.x * blockDim.x + threadIdx.x;   // r = b*nodes + node
    int R = BATCH * nodes;
    if (r >= R) return;
    int b  = r / nodes;
    int nd = r % nodes;
    int q  = op_ids[b * 63 + (nodes - 1) + nd];
    int slot = atomicAdd(&counts[q], 1);
    rowidx[q * RCAP + slot] = r;
}

// 32-row tile granularity
__global__ void bin_scan(const int* __restrict__ counts, int* __restrict__ tileOff) {
    if (threadIdx.x == 0 && blockIdx.x == 0) {
        int acc = 0;
        for (int q = 0; q < KOPS; ++q) {
            tileOff[q] = acc;
            acc += (counts[q] + 31) >> 5;
        }
        tileOff[KOPS] = acc;
    }
}

// ---------------------------------------------------------------------------
// Tree level grouped GEMM: each block = one 32-row tile of one op bin.
// Computes relu( X(32x1024) @ W_app[q](1024x512) + b_app[q] ) with WMMA bf16.
// Two K-phases of 512 columns each (32KB LDS per phase, acc live across).
// Each B-fragment feeds TWO WMMAs (two M sub-tiles) -> 32 FLOP/byte from L2.
// A staged in LDS in the ISA 16-bit A 16x32 per-lane layout:
//   lane L: m = L&15, kbase = (L<16)?0:8,
//   elem i<8  -> k = kbase + i ; elem i>=8 -> k = 16 + kbase + (i-8)
// ---------------------------------------------------------------------------
__global__ void __launch_bounds__(128) tree_gemm(
        const unsigned short* __restrict__ encIn,
        unsigned short* __restrict__ encOut,
        const unsigned short* __restrict__ Wt,      // [16][32 cs][32 nt][512]
        const float* __restrict__ b_app,            // [16][512]
        const int* __restrict__ counts,
        const int* __restrict__ tileOff,
        const int* __restrict__ rowidx,
        int nodes) {
    __shared__ __align__(16) unsigned short lds_a[2 * 16 * 512]; // [mt][csL][lane][16]
    __shared__ int rows_s[32];

    int bid = blockIdx.x;
    if (bid >= tileOff[KOPS]) return;
    int q = 0;
    while (q < KOPS - 1 && bid >= tileOff[q + 1]) ++q;
    int localTile = bid - tileOff[q];
    int tid = threadIdx.x;

    if (tid < 32) {
        int cnt = counts[q];
        int rr  = localTile * 32 + tid;
        rows_s[tid] = (rr < cnt) ? rowidx[q * RCAP + rr] : -1;
    }

    int wave = tid >> 5;
    int lane = tid & 31;

    f32x8 acc[2][8];
    f32x8 zero = {};
#pragma unroll
    for (int mt = 0; mt < 2; ++mt)
#pragma unroll
        for (int j = 0; j < 8; ++j) acc[mt][j] = zero;

    for (int ph = 0; ph < 2; ++ph) {
        __syncthreads();   // also covers rows_s on first iteration
        // Fill A LDS: 2 mt x 16 csL x 32 lanes x 2 halves = 2048 chunks (16B ea).
        for (int ch = tid; ch < 2048; ch += 128) {
            int half  = ch & 1;
            int lane2 = (ch >> 1) & 31;
            int csL   = (ch >> 6) & 15;
            int mt    = ch >> 10;
            int m     = lane2 & 15;
            int kb    = (lane2 < 16) ? 0 : 8;
            int c0    = (ph * 16 + csL) * 32 + (half ? 16 : 0) + kb;
            int r     = rows_s[mt * 16 + m];
            uint4 val = make_uint4(0u, 0u, 0u, 0u);
            if (r >= 0) {
                int b  = r / nodes;
                int nd = r % nodes;
                // phase 0 -> child 0 (c0 < 512); phase 1 -> child 1
                int coff = c0 - ph * HDIM;
                long crow = (long)(b * 2 * nodes + 2 * nd + ph);
                val = *(const uint4*)(encIn + crow * HDIM + coff);
            }
            *(uint4*)&lds_a[mt * 8192 + csL * 512 + lane2 * 16 + half * 8] = val;
        }
        __syncthreads();

        for (int csL = 0; csL < 16; ++csL) {
            int cs = ph * 16 + csL;
            bf16x16 a0 = *(const bf16x16*)&lds_a[csL * 512 + lane * 16];
            bf16x16 a1 = *(const bf16x16*)&lds_a[8192 + csL * 512 + lane * 16];
#pragma unroll
            for (int j = 0; j < 8; ++j) {
                int nt = wave + 4 * j;
                bf16x16 bfr = *(const bf16x16*)(
                    Wt + ((((long)q * 32 + cs) * 32 + nt) << 9) + lane * 16);
                acc[0][j] = __builtin_amdgcn_wmma_f32_16x16x32_bf16(
                    false, a0, false, bfr, (short)0, acc[0][j], false, false);
                acc[1][j] = __builtin_amdgcn_wmma_f32_16x16x32_bf16(
                    false, a1, false, bfr, (short)0, acc[1][j], false, false);
            }
        }
    }

    int ncol = lane & 15;
#pragma unroll
    for (int mt = 0; mt < 2; ++mt) {
#pragma unroll
        for (int j = 0; j < 8; ++j) {
            int nt  = wave + 4 * j;
            int col = nt * 16 + ncol;
            float bias = b_app[q * HDIM + col];
#pragma unroll
            for (int v = 0; v < 8; ++v) {
                int m = (lane < 16) ? v : (v + 8);
                int r = rows_s[mt * 16 + m];
                if (r >= 0) {
                    float val = fmaxf(acc[mt][j][v] + bias, 0.0f);
                    encOut[(long)r * HDIM + col] = f32_to_bf16(val);
                }
            }
        }
    }
}

// ---------------------------------------------------------------------------
// Dense GEMM: relu( A(M x C) @ Wt(C x 512) + bias ), A bf16 row-major, out bf16.
// 32-row M tiles, K phases of 512 columns (C must be a multiple of 512).
// ---------------------------------------------------------------------------
__global__ void __launch_bounds__(128) dense_gemm(
        const unsigned short* __restrict__ A, int lda,
        const unsigned short* __restrict__ Wt,      // [C/32][32][512]
        const float* __restrict__ bias,
        unsigned short* __restrict__ out, int ldout,
        int M, int C) {
    __shared__ __align__(16) unsigned short lds_a[2 * 16 * 512];
    int tid  = threadIdx.x;
    int row0 = blockIdx.x * 32;
    int nPh  = C >> 9;

    int wave = tid >> 5;
    int lane = tid & 31;

    f32x8 acc[2][8];
    f32x8 zero = {};
#pragma unroll
    for (int mt = 0; mt < 2; ++mt)
#pragma unroll
        for (int j = 0; j < 8; ++j) acc[mt][j] = zero;

    for (int ph = 0; ph < nPh; ++ph) {
        __syncthreads();
        for (int ch = tid; ch < 2048; ch += 128) {
            int half  = ch & 1;
            int lane2 = (ch >> 1) & 31;
            int csL   = (ch >> 6) & 15;
            int mt    = ch >> 10;
            int m     = lane2 & 15;
            int kb    = (lane2 < 16) ? 0 : 8;
            int c0    = (ph * 16 + csL) * 32 + (half ? 16 : 0) + kb;
            int r     = row0 + mt * 16 + m;
            uint4 val = make_uint4(0u, 0u, 0u, 0u);
            if (r < M) val = *(const uint4*)(A + (long)r * lda + c0);
            *(uint4*)&lds_a[mt * 8192 + csL * 512 + lane2 * 16 + half * 8] = val;
        }
        __syncthreads();

        for (int csL = 0; csL < 16; ++csL) {
            int cs = ph * 16 + csL;
            bf16x16 a0 = *(const bf16x16*)&lds_a[csL * 512 + lane * 16];
            bf16x16 a1 = *(const bf16x16*)&lds_a[8192 + csL * 512 + lane * 16];
#pragma unroll
            for (int j = 0; j < 8; ++j) {
                int nt = wave + 4 * j;
                bf16x16 bfr = *(const bf16x16*)(
                    Wt + (((long)cs * 32 + nt) << 9) + lane * 16);
                acc[0][j] = __builtin_amdgcn_wmma_f32_16x16x32_bf16(
                    false, a0, false, bfr, (short)0, acc[0][j], false, false);
                acc[1][j] = __builtin_amdgcn_wmma_f32_16x16x32_bf16(
                    false, a1, false, bfr, (short)0, acc[1][j], false, false);
            }
        }
    }

    int ncol = lane & 15;
#pragma unroll
    for (int mt = 0; mt < 2; ++mt) {
#pragma unroll
        for (int j = 0; j < 8; ++j) {
            int nt  = wave + 4 * j;
            int col = nt * 16 + ncol;
            float bv = bias[col];
#pragma unroll
            for (int v = 0; v < 8; ++v) {
                int m = (lane < 16) ? v : (v + 8);
                int r = row0 + mt * 16 + m;
                if (r < M) {
                    float val = fmaxf(acc[mt][j][v] + bv, 0.0f);
                    out[(long)r * ldout + col] = f32_to_bf16(val);
                }
            }
        }
    }
}

// Concatenate task_features (fp32 -> bf16) into z[:, 512:1024]
__global__ void concat_task(const float* __restrict__ tf,
                            unsigned short* __restrict__ z) {
    int idx = blockIdx.x * blockDim.x + threadIdx.x;   // B*H
    if (idx >= BATCH * HDIM) return;
    int b = idx / HDIM;
    int c = idx % HDIM;
    z[b * 1024 + 512 + c] = f32_to_bf16(tf[idx]);
}

// Head: out[b] = softplus( h[b,:] . W_d2 + b_d2 )
__global__ void head_gemv(const unsigned short* __restrict__ h,
                          const float* __restrict__ Wd2,
                          const float* __restrict__ bd2,
                          float* __restrict__ out) {
    __shared__ float red[128];
    int b = blockIdx.x, tid = threadIdx.x;
    float s = 0.0f;
    for (int c = tid; c < HDIM; c += 128)
        s += bf16_to_f32(h[(long)b * HDIM + c]) * Wd2[c];
    red[tid] = s;
    __syncthreads();
    for (int off = 64; off > 0; off >>= 1) {
        if (tid < off) red[tid] += red[tid + off];
        __syncthreads();
    }
    if (tid == 0) {
        float x = red[0] + bd2[0];
        out[b] = fmaxf(x, 0.0f) + log1pf(expf(-fabsf(x)));   // stable softplus
    }
}

// ---------------------------------------------------------------------------
extern "C" void kernel_launch(void* const* d_in, const int* in_sizes, int n_in,
                              void* d_out, int out_size, void* d_ws, size_t ws_size,
                              hipStream_t stream) {
    const float* task_features = (const float*)d_in[0];   // (B, H)
    const float* leaf_table    = (const float*)d_in[1];   // (32, H)
    const float* W_app         = (const float*)d_in[2];   // (16, 1024, 512)
    const float* b_app         = (const float*)d_in[3];   // (16, 512)
    const float* W_lam         = (const float*)d_in[4];   // (512, 512)
    const float* b_lam         = (const float*)d_in[5];   // (512,)
    const float* W_d1          = (const float*)d_in[6];   // (1024, 512)
    const float* b_d1          = (const float*)d_in[7];   // (512,)
    const float* W_d2          = (const float*)d_in[8];   // (512, 1)
    const float* b_d2          = (const float*)d_in[9];   // (1,)
    const int*   leaf_ids      = (const int*)d_in[10];    // (B, 64)
    const int*   op_ids        = (const int*)d_in[11];    // (B, 63)
    float*       out           = (float*)d_out;

    // Workspace layout
    char*  ws  = (char*)d_ws;
    size_t off = 0;
    auto alloc = [&](size_t bytes) { size_t o = off; off = (off + bytes + 255) & ~(size_t)255; return o; };
    unsigned short* WtApp = (unsigned short*)(ws + alloc((size_t)16 * 1024 * 512 * 2));
    unsigned short* WtLam = (unsigned short*)(ws + alloc((size_t)512 * 512 * 2));
    unsigned short* WtD1  = (unsigned short*)(ws + alloc((size_t)1024 * 512 * 2));
    unsigned short* encA  = (unsigned short*)(ws + alloc((size_t)BATCH * 64 * HDIM * 2));
    unsigned short* encB  = (unsigned short*)(ws + alloc((size_t)BATCH * 32 * HDIM * 2));
    unsigned short* zbuf  = (unsigned short*)(ws + alloc((size_t)BATCH * 1024 * 2));
    unsigned short* hbuf  = (unsigned short*)(ws + alloc((size_t)BATCH * HDIM * 2));
    int* rowidx  = (int*)(ws + alloc((size_t)KOPS * RCAP * 4));
    int* counts  = (int*)(ws + alloc(64));
    int* tileOff = (int*)(ws + alloc(128));

    // 1) Tile weights to bf16 WMMA layout (redone every call; deterministic)
    {
        long total = (long)16 * 1024 * 512;
        tile_weights<<<(int)((total + 255) / 256), 256, 0, stream>>>(W_app, WtApp, 1024, 512, total);
    }
    {
        long total = (long)512 * 512;
        tile_weights<<<(int)((total + 255) / 256), 256, 0, stream>>>(W_lam, WtLam, 512, 512, total);
    }
    {
        long total = (long)1024 * 512;
        tile_weights<<<(int)((total + 255) / 256), 256, 0, stream>>>(W_d1, WtD1, 1024, 512, total);
    }

    // 2) Leaf gather -> encA (B x 64 x H, bf16)
    {
        long total = (long)BATCH * 64 * HDIM;
        gather_leaves<<<(int)((total + 255) / 256), 256, 0, stream>>>(leaf_table, leaf_ids, encA);
    }

    // 3) Tree levels (grouped GEMM per level, 32-row tiles)
    unsigned short* encIn  = encA;
    unsigned short* encOut = encB;
    for (int lvl = DLVL - 1; lvl >= 0; --lvl) {
        int nodes = 1 << lvl;
        int R     = BATCH * nodes;
        bin_zero<<<1, 32, 0, stream>>>(counts);
        bin_rows<<<R / 256, 256, 0, stream>>>(op_ids, nodes, counts, rowidx);
        bin_scan<<<1, 1, 0, stream>>>(counts, tileOff);
        tree_gemm<<<R / 32 + (KOPS - 1), 128, 0, stream>>>(
            encIn, encOut, WtApp, b_app, counts, tileOff, rowidx, nodes);
        unsigned short* t = encIn; encIn = encOut; encOut = t;
    }
    unsigned short* root = encIn;   // (B x 1 x H)

    // 4) sk = relu(root @ W_lam + b_lam) -> z[:, :512]
    dense_gemm<<<BATCH / 32, 128, 0, stream>>>(root, HDIM, WtLam, b_lam, zbuf, 1024, BATCH, 512);
    // 5) z[:, 512:] = task_features
    concat_task<<<(BATCH * HDIM) / 256, 256, 0, stream>>>(task_features, zbuf);
    // 6) h = relu(z @ W_d1 + b_d1)
    dense_gemm<<<BATCH / 32, 128, 0, stream>>>(zbuf, 1024, WtD1, b_d1, hbuf, 512, BATCH, 1024);
    // 7) out = softplus(h @ W_d2 + b_d2)
    head_gemv<<<BATCH, 128, 0, stream>>>(hbuf, W_d2, b_d2, out);
}
